// MultiHeadAttention_2276332667508
// MI455X (gfx1250) — matrix operand order repaired and checked by reading, not compiled, not run on hardware
//
#include <hip/hip_runtime.h>
#include <hip/hip_bf16.h>

// ---------------------------------------------------------------------------
// MI455X (gfx1250, wave32) multi-head attention forward.
// Heavy math: v_wmma_f32_16x16x32_f16 (fp32 accumulate).
// Data movement: global_load_async_to_lds_b128 (ASYNCcnt) + double buffering.
// ---------------------------------------------------------------------------

typedef __attribute__((ext_vector_type(16))) _Float16 v16h;
typedef __attribute__((ext_vector_type(8)))  float    v8f;

union HFrag { v16h v; unsigned int u[8]; };

__device__ __forceinline__ unsigned short f2h(float f) {
  union { _Float16 h; unsigned short u; } c;
  c.h = (_Float16)f;
  return c.u;
}
__device__ __forceinline__ unsigned int pack2(float a, float b) {
  return (unsigned int)f2h(a) | ((unsigned int)f2h(b) << 16);
}

// Async DMA: 16 contiguous bytes global -> 16 contiguous bytes LDS, per lane.
// VDST operand carries the LDS byte address (low 32 bits of the generic
// shared-aperture pointer == LDS offset, per ISA 10.2 aperture rules).
__device__ __forceinline__ void async_g2l_b128(void* lds, const void* gptr) {
  unsigned off = (unsigned)(size_t)lds;
  asm volatile("global_load_async_to_lds_b128 %0, %1, off"
               :: "v"(off), "v"(gptr) : "memory");
}
__device__ __forceinline__ void wait_async0() {
  asm volatile("s_wait_asynccnt 0" ::: "memory");
}

// ---------------------------------------------------------------------------
// fp32 -> packed f16 conversion (float4 in, uint2 out; n always /4)
// ---------------------------------------------------------------------------
__global__ __launch_bounds__(256)
void cvt_f32_to_f16(const float4* __restrict__ in, uint2* __restrict__ out, int n4) {
  int i = blockIdx.x * 256 + threadIdx.x;
  if (i < n4) {
    float4 f = in[i];
    uint2 o;
    o.x = pack2(f.x, f.y);
    o.y = pack2(f.z, f.w);
    out[i] = o;
  }
}

// ---------------------------------------------------------------------------
// C[M,N] = A[M,K] * Bt[N,K]^T   (A, Bt f16 row-major; torch-Linear x @ W.T)
// Block tile 128x128, K-step 32. LDS rows are 20 uints (80B): 16B-aligned for
// async B128 DMA and conflict-free for the WMMA fragment gather (20*m mod 64
// spans 16 distinct bank groups). Double-buffered: tile k+1 streams in via
// the async engine while tile k feeds 8 WMMAs/wave. One barrier per K-step.
// mode 0: fp32 out + bias.   mode 1: f16 out scattered to [B,NH,T,HD].
// ---------------------------------------------------------------------------
#define GST 20   // GEMM LDS row stride in uints

__global__ __launch_bounds__(256)
void gemm_f16_wmma(const unsigned short* __restrict__ A,
                   const unsigned short* __restrict__ Bt,
                   float* __restrict__ Cf,
                   unsigned short* __restrict__ Ch,
                   const float* __restrict__ bias,
                   int M, int N, int K, int mode, int Tseq) {
  __shared__ unsigned int As[2][128 * GST];
  __shared__ unsigned int Bs[2][128 * GST];

  const unsigned int* Au = (const unsigned int*)A;   // K/2 uints per row
  const unsigned int* Bu = (const unsigned int*)Bt;
  const int Ku = K >> 1;

  const int tid  = threadIdx.x;
  const int wave = tid >> 5, lane = tid & 31;
  const int lg = lane >> 4, l16 = lane & 15;         // lane group / lane%16
  const int m_blk = blockIdx.x * 128, n_blk = blockIdx.y * 128;
  const int wm = (wave & 3) * 32;                    // wave row offset
  const int wn = (wave >> 2) * 64;                   // wave col offset

  // This thread's two 16B DMA chunks per tile: (row, 16B-chunk-in-row)
  const int r0 = tid >> 2,           c0 = (tid & 3) * 4;          // ids 0..255
  const int r1 = (256 + tid) >> 2,   c1 = (tid & 3) * 4;          // ids 256..511

  const v8f vzero = {0.f, 0.f, 0.f, 0.f, 0.f, 0.f, 0.f, 0.f};
  v8f acc[2][4];
  for (int i = 0; i < 2; ++i)
    for (int j = 0; j < 4; ++j) acc[i][j] = vzero;

  // Prologue: DMA tile 0 into buffer 0.
  {
    async_g2l_b128(&As[0][r0 * GST + c0], Au + (size_t)(m_blk + r0) * Ku + c0);
    async_g2l_b128(&As[0][r1 * GST + c1], Au + (size_t)(m_blk + r1) * Ku + c1);
    async_g2l_b128(&Bs[0][r0 * GST + c0], Bu + (size_t)(n_blk + r0) * Ku + c0);
    async_g2l_b128(&Bs[0][r1 * GST + c1], Bu + (size_t)(n_blk + r1) * Ku + c1);
  }

  for (int k0 = 0; k0 < K; k0 += 32) {
    const int cur = (k0 >> 5) & 1;
    wait_async0();        // own DMA chunks of tile k are in LDS
    __syncthreads();      // everyone's chunks are in LDS; prev tile fully read

    if (k0 + 32 < K) {    // stream tile k+1 into the other buffer
      const int nk = (k0 + 32) >> 1;
      async_g2l_b128(&As[cur ^ 1][r0 * GST + c0], Au + (size_t)(m_blk + r0) * Ku + nk + c0);
      async_g2l_b128(&As[cur ^ 1][r1 * GST + c1], Au + (size_t)(m_blk + r1) * Ku + nk + c1);
      async_g2l_b128(&Bs[cur ^ 1][r0 * GST + c0], Bu + (size_t)(n_blk + r0) * Ku + nk + c0);
      async_g2l_b128(&Bs[cur ^ 1][r1 * GST + c1], Bu + (size_t)(n_blk + r1) * Ku + nk + c1);
    }
    if (k0 + 64 < K) {    // L2 hint two tiles ahead (global_prefetch_b8)
      __builtin_prefetch(&Au[(size_t)(m_blk + (tid >> 4)) * Ku + (k0 >> 1) + 32 + (tid & 15)], 0, 1);
      __builtin_prefetch(&Bu[(size_t)(n_blk + (tid >> 4)) * Ku + (k0 >> 1) + 32 + (tid & 15)], 0, 1);
    }

    // A fragments (ISA 16-bit A 16x32 layout): lane row = l16,
    // uints 0..3 -> K = lg*8 + {0..7}, uints 4..7 -> K = 16 + lg*8 + {0..7}
    HFrag af[2];
#pragma unroll
    for (int s = 0; s < 2; ++s) {
      int m = wm + s * 16 + l16;
#pragma unroll
      for (int p = 0; p < 4; ++p) {
        af[s].u[p]     = As[cur][m * GST + lg * 4 + p];
        af[s].u[4 + p] = As[cur][m * GST + 8 + lg * 4 + p];
      }
    }
    // B fragments (32x16): lane col = l16, uints 0..3 -> K = lg*8 + {0..7}, uints 4..7 -> K = 16 + lg*8 + {0..7}
    HFrag bf[4];
#pragma unroll
    for (int s = 0; s < 4; ++s) {
      int n = wn + s * 16 + l16;
#pragma unroll
      for (int p = 0; p < 8; ++p) bf[s].u[p] = Bs[cur][n * GST + (p >> 2) * 8 + lg * 4 + (p & 3)];
    }
#pragma unroll
    for (int i = 0; i < 2; ++i)
#pragma unroll
      for (int j = 0; j < 4; ++j)
        acc[i][j] = __builtin_amdgcn_wmma_f32_16x16x32_f16(
            false, af[i].v, false, bf[j].v, (short)0, acc[i][j], false, false);
  }

  // Epilogue. D layout: M = 8*lg + r, N = l16.
#pragma unroll
  for (int i = 0; i < 2; ++i)
#pragma unroll
    for (int j = 0; j < 4; ++j) {
      int mrow = m_blk + wm + i * 16 + lg * 8;
      int ncol = n_blk + wn + j * 16 + l16;
#pragma unroll
      for (int r = 0; r < 8; ++r) {
        float v = acc[i][j][r];
        int row = mrow + r;
        if (mode == 0) {
          Cf[(size_t)row * N + ncol] = v + (bias ? bias[ncol] : 0.0f);
        } else {  // scatter to [B, NH=16, T, HD=128] as f16
          int b = row / Tseq, t = row - b * Tseq;
          int h = ncol >> 7, d = ncol & 127;
          Ch[((((size_t)b * 16 + h) * Tseq + t) << 7) + d] = f2h(v);
        }
      }
    }
}

// ---------------------------------------------------------------------------
// Causal flash attention. Q/K/V: [B,NH,T,128] f16. One block per
// (batch-head, 128 q rows); wave w owns q rows [qb+16w, qb+16w+16).
// Per 32-key chunk: K tile DMAs straight to LDS (async b128, 68-uint rows =
// 272B, 16B-aligned), V is staged transposed, S = Q K^T (8 WMMA), online
// softmax with cross-lane row reductions, O += P V (8 WMMA).
// ---------------------------------------------------------------------------
#define KST 68   // K-tile LDS row stride in uints (128 f16 data + pad)

__global__ __launch_bounds__(256)
void flash_attn_wmma(const unsigned short* __restrict__ Q,
                     const unsigned short* __restrict__ Km,
                     const unsigned short* __restrict__ V,
                     unsigned short* __restrict__ ctx,
                     int Tseq) {
  __shared__ unsigned int Ks[32 * KST];     // 32 keys x 128 hd (+pad)
  __shared__ unsigned int VtU[128 * 17];    // V transposed: [d][k] (+pad)
  __shared__ unsigned int Pl[8][16 * 17];   // per-wave P relayout scratch
  unsigned short* Vts = (unsigned short*)VtU;

  const int tid  = threadIdx.x;
  const int wave = tid >> 5, lane = tid & 31;
  const int lg = lane >> 4, l16 = lane & 15;
  const int bh = blockIdx.x;                 // batch*NH + head
  const int qb = blockIdx.y * 128;
  const int qw = qb + wave * 16;             // this wave's first q row
  const size_t base = (size_t)bh * Tseq * 64;  // rows of 64 uints (128 f16)

  const unsigned int* Qu  = (const unsigned int*)Q + base;
  const unsigned int* KuG = (const unsigned int*)Km + base;
  const unsigned int* VuG = (const unsigned int*)V + base;

  // Resident Q A-fragments: 16 rows x 128 K, 4 fragments of 16x32.
  HFrag qf[4];
  {
    const size_t qrow = (size_t)(qw + l16) * 64;
#pragma unroll
    for (int f = 0; f < 4; ++f)
#pragma unroll
      for (int p = 0; p < 4; ++p) {
        qf[f].u[p]     = Qu[qrow + f * 16 + lg * 4 + p];
        qf[f].u[4 + p] = Qu[qrow + f * 16 + 8 + lg * 4 + p];
      }
  }

  const v8f vzero = {0.f, 0.f, 0.f, 0.f, 0.f, 0.f, 0.f, 0.f};
  v8f o[8];                      // 16 x 128 f32 output accumulator
#pragma unroll
  for (int j = 0; j < 8; ++j) o[j] = vzero;
  float m_st[8], l_st[8];        // per-row running max / sum (rows 8*lg+r)
#pragma unroll
  for (int r = 0; r < 8; ++r) { m_st[r] = -__builtin_inff(); l_st[r] = 0.f; }

  // This thread's two 16B K-chunks per tile: 32 rows x 16 chunks = 512 ids.
  const int kr0 = tid >> 4,         kc0 = (tid & 15) * 4;
  const int kr1 = (256 + tid) >> 4, kc1 = (tid & 15) * 4;

  const float sc = 0.088388347648318447f;   // 1/sqrt(128)
  const int kv_end = qb + 128;
  for (int kv0 = 0; kv0 < kv_end; kv0 += 32) {
    __syncthreads();   // previous tile fully consumed by all waves
    // K chunk: async DMA straight into LDS.
    async_g2l_b128(&Ks[kr0 * KST + kc0], KuG + (size_t)(kv0 + kr0) * 64 + kc0);
    async_g2l_b128(&Ks[kr1 * KST + kc1], KuG + (size_t)(kv0 + kr1) * 64 + kc1);
    // V chunk: staged transposed (DMA cannot transpose): 2048 uints.
#pragma unroll
    for (int i = 0; i < 8; ++i) {
      int idx = i * 256 + tid;
      int kr = idx >> 6, kc = idx & 63;
      unsigned int u = VuG[(size_t)(kv0 + kr) * 64 + kc];
      Vts[(2 * kc) * 34 + kr]     = (unsigned short)(u & 0xFFFFu);
      Vts[(2 * kc + 1) * 34 + kr] = (unsigned short)(u >> 16);
    }
    wait_async0();
    __syncthreads();

    // S = Q K^T : 16 rows x 32 keys as two 16x16 D tiles.
    v8f S0 = vzero, S1 = vzero;
#pragma unroll
    for (int f = 0; f < 4; ++f) {
      HFrag kb0, kb1;
#pragma unroll
      for (int p = 0; p < 8; ++p) {
        kb0.u[p] = Ks[l16 * KST + f * 16 + (p >> 2) * 8 + lg * 4 + (p & 3)];
        kb1.u[p] = Ks[(16 + l16) * KST + f * 16 + (p >> 2) * 8 + lg * 4 + (p & 3)];
      }
      S0 = __builtin_amdgcn_wmma_f32_16x16x32_f16(false, qf[f].v, false, kb0.v,
                                                  (short)0, S0, false, false);
      S1 = __builtin_amdgcn_wmma_f32_16x16x32_f16(false, qf[f].v, false, kb1.v,
                                                  (short)0, S1, false, false);
    }

    // Online softmax. D layout: row = 8*lg + r (16 lanes per row), col = l16.
    float alpha[8];
    unsigned short* Pw = (unsigned short*)&Pl[wave][0];
#pragma unroll
    for (int r = 0; r < 8; ++r) {
      const int qi = qw + 8 * lg + r;
      float s0 = (kv0 + l16      <= qi) ? S0[r] * sc : -__builtin_inff();
      float s1 = (kv0 + 16 + l16 <= qi) ? S1[r] * sc : -__builtin_inff();
      float mx = fmaxf(s0, s1);
      for (int off = 1; off < 16; off <<= 1) mx = fmaxf(mx, __shfl_xor(mx, off, 32));
      float mnew = fmaxf(m_st[r], mx);
      float p0 = __expf(s0 - mnew), p1 = __expf(s1 - mnew);
      float rs = p0 + p1;
      for (int off = 1; off < 16; off <<= 1) rs += __shfl_xor(rs, off, 32);
      alpha[r] = __expf(m_st[r] - mnew);
      l_st[r]  = l_st[r] * alpha[r] + rs;
      m_st[r]  = mnew;
      Pw[(8 * lg + r) * 34 + l16]      = f2h(p0);
      Pw[(8 * lg + r) * 34 + 16 + l16] = f2h(p1);
    }
#pragma unroll
    for (int j = 0; j < 8; ++j)
#pragma unroll
      for (int r = 0; r < 8; ++r) o[j][r] *= alpha[r];

    // In-wave LDS RAW: wait for the P stores before the fragment reload.
    asm volatile("s_wait_dscnt 0" ::: "memory");
    HFrag pf;
    const unsigned int* Pwu = &Pl[wave][0];
#pragma unroll
    for (int p = 0; p < 4; ++p) {
      pf.u[p]     = Pwu[l16 * 17 + lg * 4 + p];
      pf.u[4 + p] = Pwu[l16 * 17 + 8 + lg * 4 + p];
    }
    // O += P[16x32] * V[32x128]; B fragments read transposed V (k-pairs/uint).
#pragma unroll
    for (int j = 0; j < 8; ++j) {
      HFrag vb;
      const int d = j * 16 + l16;
#pragma unroll
      for (int p = 0; p < 8; ++p) vb.u[p] = VtU[d * 17 + (p >> 2) * 8 + lg * 4 + (p & 3)];
      o[j] = __builtin_amdgcn_wmma_f32_16x16x32_f16(false, pf.v, false, vb.v,
                                                    (short)0, o[j], false, false);
    }
  }

  // Normalize and write ctx as f16 [B*T, 2048] (heads re-interleaved).
  const int b = bh >> 4, h = bh & 15;
#pragma unroll
  for (int r = 0; r < 8; ++r) {
    const int t = qw + 8 * lg + r;
    const float inv_l = 1.0f / l_st[r];
    unsigned short* crow = ctx + ((size_t)b * Tseq + t) * 2048 + h * 128;
#pragma unroll
    for (int j = 0; j < 8; ++j)
      crow[j * 16 + l16] = f2h(o[j][r] * inv_l);
  }
}

// ---------------------------------------------------------------------------
// Orchestration (all on `stream`; workspace-only scratch, graph-capture safe)
// ---------------------------------------------------------------------------
extern "C" void kernel_launch(void* const* d_in, const int* in_sizes, int n_in,
                              void* d_out, int out_size, void* d_ws, size_t ws_size,
                              hipStream_t stream) {
  (void)in_sizes; (void)n_in; (void)out_size; (void)ws_size;
  const float* x  = (const float*)d_in[0];
  const float* Wq = (const float*)d_in[1];
  const float* Wk = (const float*)d_in[2];
  const float* Wv = (const float*)d_in[3];
  const float* Wo = (const float*)d_in[4];
  const float* bo = (const float*)d_in[5];
  float* out = (float*)d_out;

  const int B = 4, T = 2048, D = 2048;
  const size_t BT = (size_t)B * T;   // 8192 rows

  // Workspace layout (~192 MiB total, f16 everywhere):
  char* p = (char*)d_ws;
  unsigned short* xh  = (unsigned short*)p; p += BT * D * 2;          // 32 MiB
  unsigned short* wqh = (unsigned short*)p; p += (size_t)D * D * 2;   // 8 MiB
  unsigned short* wkh = (unsigned short*)p; p += (size_t)D * D * 2;
  unsigned short* wvh = (unsigned short*)p; p += (size_t)D * D * 2;
  unsigned short* woh = (unsigned short*)p; p += (size_t)D * D * 2;
  unsigned short* qh  = (unsigned short*)p; p += BT * D * 2;          // [B,NH,T,HD]
  unsigned short* kh  = (unsigned short*)p; p += BT * D * 2;
  unsigned short* vh  = (unsigned short*)p; p += BT * D * 2;
  unsigned short* ch  = (unsigned short*)p; p += BT * D * 2;          // ctx

  {
    int n4 = (int)(BT * D / 4);
    cvt_f32_to_f16<<<(n4 + 255) / 256, 256, 0, stream>>>((const float4*)x, (uint2*)xh, n4);
  }
  {
    int n4 = D * D / 4;
    cvt_f32_to_f16<<<(n4 + 255) / 256, 256, 0, stream>>>((const float4*)Wq, (uint2*)wqh, n4);
    cvt_f32_to_f16<<<(n4 + 255) / 256, 256, 0, stream>>>((const float4*)Wk, (uint2*)wkh, n4);
    cvt_f32_to_f16<<<(n4 + 255) / 256, 256, 0, stream>>>((const float4*)Wv, (uint2*)wvh, n4);
    cvt_f32_to_f16<<<(n4 + 255) / 256, 256, 0, stream>>>((const float4*)Wo, (uint2*)woh, n4);
  }

  dim3 gdim((unsigned)(BT / 128), (unsigned)(D / 128));
  // Q/K/V projections, scattered to per-head layout:
  gemm_f16_wmma<<<gdim, 256, 0, stream>>>(xh, wqh, nullptr, qh, nullptr, (int)BT, D, D, 1, T);
  gemm_f16_wmma<<<gdim, 256, 0, stream>>>(xh, wkh, nullptr, kh, nullptr, (int)BT, D, D, 1, T);
  gemm_f16_wmma<<<gdim, 256, 0, stream>>>(xh, wvh, nullptr, vh, nullptr, (int)BT, D, D, 1, T);

  // Causal flash attention, 64 batch-heads x 16 q-row blocks:
  flash_attn_wmma<<<dim3(64, (unsigned)(T / 128)), 256, 0, stream>>>(qh, kh, vh, ch, T);

  // Output projection + bias, fp32 result:
  gemm_f16_wmma<<<gdim, 256, 0, stream>>>(ch, woh, out, nullptr, bo, (int)BT, D, D, 0, T);
}
